// GoSpecificWattiPooling_79886391705810
// MI455X (gfx1250) — matrix operand order repaired and checked by reading, not compiled, not run on hardware
//
#include <hip/hip_runtime.h>

// Problem constants (from reference)
#define B_   8
#define L_   4096
#define DH_  1024
#define T_   1024
#define DG_  768
#define P_   256

typedef _Float16 v16h __attribute__((ext_vector_type(16)));
typedef _Float16 v8h  __attribute__((ext_vector_type(8)));
typedef _Float16 v4h  __attribute__((ext_vector_type(4)));
typedef float    v8f  __attribute__((ext_vector_type(8)));

static __device__ __forceinline__ v8f wmma_f16(v16h a, v16h b, v8f c) {
  // D = A(16x32 f16) * B(32x16 f16) + C(16x16 f32)
  return __builtin_amdgcn_wmma_f32_16x16x32_f16(false, a, false, b, (short)0, c,
                                                false, false);
}

// ---------------------------------------------------------------------------
// Kernel 1: H (f32) -> Hh (f16), row-major [B,L,DH]
// ---------------------------------------------------------------------------
__global__ void gswp_hconv(const float4* __restrict__ H4, v4h* __restrict__ Hh4,
                           int n4) {
  int i = blockIdx.x * 256 + threadIdx.x;
  if (i >= n4) return;
  float4 v = H4[i];
  v4h o;
  o[0] = (_Float16)v.x; o[1] = (_Float16)v.y;
  o[2] = (_Float16)v.z; o[3] = (_Float16)v.w;
  Hh4[i] = o;
}

// ---------------------------------------------------------------------------
// Kernel 2: K projection.  Kt[b][p][l] = sum_d H[b,l,d] * Wk[p,d]  (f16, P-major)
// One WG = 16 rows of L, 8 waves cover P=256 (32 cols each).
// ---------------------------------------------------------------------------
__global__ __launch_bounds__(256) void gswp_kproj(const float* __restrict__ H,
                                                  const float* __restrict__ Wk,
                                                  _Float16* __restrict__ Kt) {
  const int b    = blockIdx.x >> 8;          // L/16 = 256 blocks per batch
  const int l0   = (blockIdx.x & 255) << 4;
  const int wave = threadIdx.x >> 5;
  const int lane = threadIdx.x & 31;
  const int lo = lane & 15, hi = lane >> 4;
  const int p0 = wave * 32;

  const float* Hrow = H + ((size_t)b * L_ + l0 + lo) * DH_;

  v8f acc0, acc1;
#pragma unroll
  for (int i = 0; i < 8; ++i) { acc0[i] = 0.f; acc1[i] = 0.f; }

  for (int kc = 0; kc < DH_ / 32; ++kc) {
    const int cb = kc * 32 + hi * 8;
    v8f alo = *(const v8f*)(Hrow + cb);
    v8f ahi = *(const v8f*)(Hrow + cb + 16);
    v16h a;
#pragma unroll
    for (int i = 0; i < 8; ++i) { a[i] = (_Float16)alo[i]; a[8 + i] = (_Float16)ahi[i]; }
    v16h b0, b1;
#pragma unroll
    for (int n = 0; n < 16; ++n) {   // B(k, n=lo) = Wk[p0+lo][kc*32+k], k = 8*hi + (n&7) + 16*(n>>3)
      b0[n] = (_Float16)Wk[(size_t)(p0 + lo) * DH_ + kc * 32 + hi * 8 + (n & 7) + ((n >> 3) << 4)];
      b1[n] = (_Float16)Wk[(size_t)(p0 + 16 + lo) * DH_ + kc * 32 + hi * 8 + (n & 7) + ((n >> 3) << 4)];
    }
    acc0 = wmma_f16(a, b0, acc0);
    acc1 = wmma_f16(a, b1, acc1);
  }
  // Store transposed: row m = local l, col n = local p
#pragma unroll
  for (int v = 0; v < 8; ++v) {
    const int l = l0 + v + hi * 8;
    Kt[((size_t)b * P_ + p0 + lo) * L_ + l]      = (_Float16)acc0[v];
    Kt[((size_t)b * P_ + p0 + 16 + lo) * L_ + l] = (_Float16)acc1[v];
  }
}

// ---------------------------------------------------------------------------
// Kernel 3: Q projection.  Qh[b][t][p] = (sum_d G[b,t,d]*Wq[p,d]) * P^-0.5 (f16)
// ---------------------------------------------------------------------------
__global__ __launch_bounds__(256) void gswp_qproj(const float* __restrict__ G,
                                                  const float* __restrict__ Wq,
                                                  _Float16* __restrict__ Qh) {
  const int b    = blockIdx.x >> 6;          // T/16 = 64 blocks per batch
  const int t0   = (blockIdx.x & 63) << 4;
  const int wave = threadIdx.x >> 5;
  const int lane = threadIdx.x & 31;
  const int lo = lane & 15, hi = lane >> 4;
  const int p0 = wave * 32;

  const float* Grow = G + ((size_t)b * T_ + t0 + lo) * DG_;

  v8f acc0, acc1;
#pragma unroll
  for (int i = 0; i < 8; ++i) { acc0[i] = 0.f; acc1[i] = 0.f; }

  for (int kc = 0; kc < DG_ / 32; ++kc) {
    const int cb = kc * 32 + hi * 8;
    v8f alo = *(const v8f*)(Grow + cb);
    v8f ahi = *(const v8f*)(Grow + cb + 16);
    v16h a;
#pragma unroll
    for (int i = 0; i < 8; ++i) { a[i] = (_Float16)alo[i]; a[8 + i] = (_Float16)ahi[i]; }
    v16h b0, b1;
#pragma unroll
    for (int n = 0; n < 16; ++n) {
      b0[n] = (_Float16)Wq[(size_t)(p0 + lo) * DG_ + kc * 32 + hi * 8 + (n & 7) + ((n >> 3) << 4)];
      b1[n] = (_Float16)Wq[(size_t)(p0 + 16 + lo) * DG_ + kc * 32 + hi * 8 + (n & 7) + ((n >> 3) << 4)];
    }
    acc0 = wmma_f16(a, b0, acc0);
    acc1 = wmma_f16(a, b1, acc1);
  }
  const float scale = 0.0625f;  // P^-0.5, folded into Q
  _Float16* orow = Qh + ((size_t)b * T_ + t0) * P_;
#pragma unroll
  for (int v = 0; v < 8; ++v) {
    const int t = v + hi * 8;
    orow[(size_t)t * P_ + p0 + lo]      = (_Float16)(acc0[v] * scale);
    orow[(size_t)t * P_ + p0 + 16 + lo] = (_Float16)(acc1[v] * scale);
  }
}

// ---------------------------------------------------------------------------
// Kernel 4: streaming-softmax attention.
// Grid: B*(T/16) workgroups, 256 threads (8 waves).
// WG owns 16 query rows; wave w owns DH slice [128w,128w+128) of Z,
// and computes S column tile [16w,16w+16) of each 128-wide L chunk.
// ---------------------------------------------------------------------------
#define LCHUNK 128
#define PPITCH 136  // 128 + 8 halves pad -> 272B row pitch, 16B aligned

__global__ __launch_bounds__(256) void gswp_attn(const _Float16* __restrict__ Qh,
                                                 const _Float16* __restrict__ Kt,
                                                 const _Float16* __restrict__ Hh,
                                                 float* __restrict__ Out) {
  __shared__ __align__(16) _Float16 probs[2][16 * PPITCH];
  __shared__ float smax[2][8][16];
  __shared__ float ssum[2][8][16];

  const int b    = blockIdx.x >> 6;          // T/16 = 64
  const int t0   = (blockIdx.x & 63) << 4;
  const int wave = threadIdx.x >> 5;
  const int lane = threadIdx.x & 31;
  const int lo = lane & 15, hi = lane >> 4;
  const int d0 = wave << 7;

  // Resident Q fragments: 16 rows x 256 (A layout, 8 k-chunks of 32)
  v16h qf[8];
  {
    const _Float16* qrow = Qh + ((size_t)b * T_ + t0 + lo) * P_;
#pragma unroll
    for (int kc = 0; kc < 8; ++kc) {
      v8h a_lo = *(const v8h*)(qrow + kc * 32 + hi * 8);
      v8h a_hi = *(const v8h*)(qrow + kc * 32 + hi * 8 + 16);
#pragma unroll
      for (int i = 0; i < 8; ++i) { qf[kc][i] = a_lo[i]; qf[kc][8 + i] = a_hi[i]; }
    }
  }

  float mrun[8], srun[8];
  v8f acc[8];
#pragma unroll
  for (int v = 0; v < 8; ++v) {
    mrun[v] = -1e30f; srun[v] = 0.f;
#pragma unroll
    for (int i = 0; i < 8; ++i) acc[v][i] = 0.f;
  }

  const _Float16* KtB = Kt + (size_t)b * P_ * L_;
  const _Float16* HhB = Hh + (size_t)b * L_ * DH_;

  for (int l0c = 0; l0c < L_; l0c += LCHUNK) {
    const int buf = (l0c >> 7) & 1;

    // ---- 1) S column tile for this wave (cols l0c + 16*wave .. +15) ----
    v8f s;
#pragma unroll
    for (int i = 0; i < 8; ++i) s[i] = 0.f;
#pragma unroll
    for (int kc = 0; kc < 8; ++kc) {
      // B(k, n=lo) = Kt[b][kc*32+k][l0c + 16*wave + lo], k = 8*hi + (i&7) + 16*(i>>3)
      v16h bf;
#pragma unroll
      for (int i = 0; i < 16; ++i)
        bf[i] = KtB[(size_t)(kc * 32 + hi * 8 + (i & 7) + ((i >> 3) << 4)) * L_ + l0c + wave * 16 + lo];
      s = wmma_f16(qf[kc], bf, s);
    }

    // ---- 2) per-row max over this wave's 16 columns ----
    float mv[8];
#pragma unroll
    for (int v = 0; v < 8; ++v) {
      float m = s[v];
      m = fmaxf(m, __shfl_xor(m, 1, 32));
      m = fmaxf(m, __shfl_xor(m, 2, 32));
      m = fmaxf(m, __shfl_xor(m, 4, 32));
      m = fmaxf(m, __shfl_xor(m, 8, 32));
      mv[v] = m;
    }
    if (lo == 0) {
#pragma unroll
      for (int v = 0; v < 8; ++v) smax[buf][wave][v + hi * 8] = mv[v];
    }
    __syncthreads();  // B1: wave maxima visible

    // ---- 3) combine max, rescale factor, exp, partial sums, probs->LDS ----
    float alpha[8];
#pragma unroll
    for (int v = 0; v < 8; ++v) {
      const int r = v + hi * 8;
      float cm = smax[buf][0][r];
#pragma unroll
      for (int w = 1; w < 8; ++w) cm = fmaxf(cm, smax[buf][w][r]);
      const float mnew = fmaxf(mrun[v], cm);
      alpha[v] = __expf(mrun[v] - mnew);
      mrun[v] = mnew;
      const float p = __expf(s[v] - mnew);
      probs[buf][(size_t)r * PPITCH + wave * 16 + lo] = (_Float16)p;
      float rs = p;
      rs += __shfl_xor(rs, 1, 32);
      rs += __shfl_xor(rs, 2, 32);
      rs += __shfl_xor(rs, 4, 32);
      rs += __shfl_xor(rs, 8, 32);
      if (lo == 0) ssum[buf][wave][r] = rs;
    }
    __syncthreads();  // B2: probs + wave sums visible

    // ---- 4) running-sum update, accumulator rescale, Z += P~ @ H ----
#pragma unroll
    for (int v = 0; v < 8; ++v) {
      const int r = v + hi * 8;
      float cs = 0.f;
#pragma unroll
      for (int w = 0; w < 8; ++w) cs += ssum[buf][w][r];
      srun[v] = srun[v] * alpha[v] + cs;
    }
#pragma unroll
    for (int ct = 0; ct < 8; ++ct)
#pragma unroll
      for (int v = 0; v < 8; ++v) acc[ct][v] *= alpha[v];

#pragma unroll
    for (int kc = 0; kc < 4; ++kc) {
      // A fragment of probs (16 x 32 over l), read from row-major [row][l] LDS
      v8h p_lo = *(const v8h*)&probs[buf][(size_t)lo * PPITCH + kc * 32 + hi * 8];
      v8h p_hi = *(const v8h*)&probs[buf][(size_t)lo * PPITCH + kc * 32 + hi * 8 + 16];
      v16h pa;
#pragma unroll
      for (int i = 0; i < 8; ++i) { pa[i] = p_lo[i]; pa[8 + i] = p_hi[i]; }
      const _Float16* hrow = HhB + (size_t)(l0c + kc * 32 + hi * 8) * DH_ + d0 + lo;
#pragma unroll
      for (int ct = 0; ct < 8; ++ct) {
        v16h hb;  // B(k, n=lo)=Hh[l0c+kc*32+k, d0+16ct+lo], k = 8*hi + (i&7) + 16*(i>>3)
#pragma unroll
        for (int i = 0; i < 16; ++i)
          hb[i] = hrow[(size_t)((i & 7) + ((i >> 3) << 4)) * DH_ + ct * 16];
        acc[ct] = wmma_f16(pa, hb, acc[ct]);
      }
    }
  }

  // ---- finalize: Z / (s + eps) ----
  float inv[8];
#pragma unroll
  for (int v = 0; v < 8; ++v) inv[v] = 1.f / (srun[v] + 1e-8f);
  float* orow = Out + ((size_t)b * T_ + t0) * DH_;
#pragma unroll
  for (int ct = 0; ct < 8; ++ct)
#pragma unroll
    for (int v = 0; v < 8; ++v)
      orow[(size_t)(v + hi * 8) * DH_ + d0 + ct * 16 + lo] = acc[ct][v] * inv[v];
}

// ---------------------------------------------------------------------------
extern "C" void kernel_launch(void* const* d_in, const int* in_sizes, int n_in,
                              void* d_out, int out_size, void* d_ws, size_t ws_size,
                              hipStream_t stream) {
  const float* H  = (const float*)d_in[0];  // [B,L,DH]
  const float* G  = (const float*)d_in[1];  // [B,T,DG]
  const float* Wq = (const float*)d_in[2];  // [P,DG]
  const float* Wk = (const float*)d_in[3];  // [P,DH]
  float* Out = (float*)d_out;               // [B,T,DH] f32

  char* ws = (char*)d_ws;
  _Float16* Hh = (_Float16*)ws;                                   // 64 MiB
  _Float16* Kt = (_Float16*)(ws + (size_t)67108864);              // 16 MiB
  _Float16* Qh = (_Float16*)(ws + (size_t)67108864 + 16777216);   // 4 MiB

  {
    const int n4 = B_ * L_ * DH_ / 4;
    gswp_hconv<<<(n4 + 255) / 256, 256, 0, stream>>>((const float4*)H, (v4h*)Hh, n4);
  }
  gswp_kproj<<<B_ * (L_ / 16), 256, 0, stream>>>(H, Wk, Kt);
  gswp_qproj<<<B_ * (T_ / 16), 256, 0, stream>>>(G, Wq, Qh);
  gswp_attn <<<B_ * (T_ / 16), 256, 0, stream>>>(Qh, Kt, Hh, Out);
}
